// GATLayer_1_21964462752234
// MI455X (gfx1250) — matrix operand order repaired and checked by reading, not compiled, not run on hardware
//
#include <hip/hip_runtime.h>
#include <math.h>

// ---------------- problem constants (match reference) ----------------
#define NN   50000     // nodes
#define NE   800000    // edges
#define INF1 128       // layer-1 input features
#define F1   128       // layer-1 output features (HEADS*HID)
#define H1   2         // layer-1 heads
#define C1   64        // layer-1 per-head channels
#define F2   32        // layer-2 output features (1 head)

// ---------------- WMMA types ----------------
typedef __attribute__((ext_vector_type(16))) __bf16 v16bf;
typedef __attribute__((ext_vector_type(8)))  float  v8f;

// ---------------- helpers ----------------
__device__ __forceinline__ unsigned enc_f(float f) {
  unsigned u = __float_as_uint(f);
  return (u & 0x80000000u) ? ~u : (u | 0x80000000u);
}
__device__ __forceinline__ float dec_f(unsigned u) {
  return (u & 0x80000000u) ? __uint_as_float(u ^ 0x80000000u) : __uint_as_float(~u);
}
__device__ __forceinline__ float lrelu(float x) { return x > 0.f ? x : 0.2f * x; }

// A fragment: 16x32 bf16, row-major source (lda = K-stride)
__device__ __forceinline__ v16bf load_a_frag(const float* __restrict__ A, int lda,
                                             int m0, int k0, int lane) {
  int half = lane >> 4;
  const float* row = A + (size_t)(m0 + (lane & 15)) * lda + k0;
  v16bf a;
#pragma unroll
  for (int g = 0; g < 8; ++g) {
    int kb = (g < 4 ? 2 * g : 16 + 2 * (g - 4)) + 8 * half;
    a[2 * g]     = (__bf16)row[kb];
    a[2 * g + 1] = (__bf16)row[kb + 1];
  }
  return a;
}
// B fragment: 32x16 bf16, row-major source (ldb = N-stride)
__device__ __forceinline__ v16bf load_b_frag(const float* __restrict__ B, int ldb,
                                             int k0, int n0, int lane) {
  int half = lane >> 4;
  int n = n0 + (lane & 15);
  v16bf b;
#pragma unroll
  for (int g = 0; g < 8; ++g) {
    int k = k0 + (g < 4 ? 2 * g : 16 + 2 * (g - 4)) + 8 * half;
    b[2 * g]     = (__bf16)B[(size_t)k * ldb + n];
    b[2 * g + 1] = (__bf16)B[(size_t)(k + 1) * ldb + n];
  }
  return b;
}

// ---------------- GEMM: C[M,NB] = A[M,K] @ B[K,NB], bf16 WMMA, f32 acc --------
// Each wave computes a 32x16 C tile (two 16x16 WMMA tiles sharing one B frag).
// grid.x = ceil(M/32); blockDim.x = 32*(NB/16) (one wave per 16-col tile).
// K, NB compile-time -> fully unrolled, 8 static WMMAs per wave.
template <int K, int NB>
__global__ void k_gemm_bf16(const float* __restrict__ A, const float* __restrict__ B,
                            float* __restrict__ C, int M) {
  int lane = threadIdx.x & 31;
  int wave = threadIdx.x >> 5;
  int m0 = blockIdx.x * 32;
  int n0 = wave * 16;
  if (m0 >= M) return;
  const bool two = (m0 + 32) <= M;  // uniform per block: second M-tile in range?
  v8f acc0 = {}, acc1 = {};
#pragma unroll
  for (int k0 = 0; k0 < K; k0 += 32) {
    v16bf b = load_b_frag(B, NB, k0, n0, lane);
    v16bf a0 = load_a_frag(A, K, m0, k0, lane);
    acc0 = __builtin_amdgcn_wmma_f32_16x16x32_bf16(false, a0, false, b,
                                                   (short)0, acc0, false, false);
    if (two) {
      v16bf a1 = load_a_frag(A, K, m0 + 16, k0, lane);
      acc1 = __builtin_amdgcn_wmma_f32_16x16x32_bf16(false, a1, false, b,
                                                     (short)0, acc1, false, false);
    }
  }
  int half = lane >> 4, n = lane & 15;
#pragma unroll
  for (int g = 0; g < 8; ++g)
    C[(size_t)(m0 + g + 8 * half) * NB + n0 + n] = acc0[g];
  if (two) {
#pragma unroll
    for (int g = 0; g < 8; ++g)
      C[(size_t)(m0 + 16 + g + 8 * half) * NB + n0 + n] = acc1[g];
  }
}

// ---------------- graph kernels ----------------
__global__ void k_deg_loop(const int* __restrict__ src, const int* __restrict__ dst,
                           const float* __restrict__ ew, float* deg, float* lsum, int E) {
  int e = blockIdx.x * blockDim.x + threadIdx.x;
  if (e >= E) return;
  int d = dst[e];
  atomicAdd(&deg[d], 1.f);
  atomicAdd(&lsum[d], ew[e]);
}

__global__ void k_loop_attr(float* lattr, const float* deg, int n) {
  int i = blockIdx.x * blockDim.x + threadIdx.x;
  if (i < n) lattr[i] = lattr[i] / fmaxf(deg[i], 1.f);
}

// kc[0..1] = sum_c We1[h*64+c]*att_edge1[h*64+c]; kc[2] = sum_c We2[c]*att_edge2[c]
__global__ void k_consts(const float* We1, const float* ae1,
                         const float* We2, const float* ae2, float* kc) {
  if (threadIdx.x | blockIdx.x) return;
  for (int h = 0; h < H1; ++h) {
    float s = 0.f;
    for (int c = 0; c < C1; ++c) s += We1[h * C1 + c] * ae1[h * C1 + c];
    kc[h] = s;
  }
  float s = 0.f;
  for (int c = 0; c < F2; ++c) s += We2[c] * ae2[c];
  kc[2] = s;
}

__global__ void k_node_attn(const float* __restrict__ h, const float* __restrict__ asv,
                            const float* __restrict__ adv, float* as_, float* ad_,
                            int n, int heads, int C) {
  int idx = blockIdx.x * blockDim.x + threadIdx.x;
  if (idx >= n * heads) return;
  int node = idx / heads, hd = idx % heads;
  const float* row = h + (size_t)node * heads * C + hd * C;
  float s = 0.f, d = 0.f;
  for (int c = 0; c < C; ++c) {
    float v = row[c];
    s += v * asv[hd * C + c];
    d += v * adv[hd * C + c];
  }
  as_[idx] = s;
  ad_[idx] = d;
}

// alpha = leaky_relu(a_src[src]+a_dst[dst]+ew*kc); store alpha, atomicMax per dst
__global__ void k_edge_alpha_max(const int* __restrict__ src, const int* __restrict__ dst,
                                 const float* __restrict__ ew, const float* __restrict__ as_,
                                 const float* __restrict__ ad_, const float* kc, int kcoff,
                                 float* alpha, unsigned* amax_enc, int E, int heads) {
  int idx = blockIdx.x * blockDim.x + threadIdx.x;
  if (idx >= E * heads) return;
  int e = idx / heads, hd = idx % heads;
  float al = lrelu(as_[src[e] * heads + hd] + ad_[dst[e] * heads + hd] + ew[e] * kc[kcoff + hd]);
  alpha[idx] = al;
  atomicMax(&amax_enc[dst[e] * heads + hd], enc_f(al));
}

// per node: self-loop alpha, final max, and self exp contribution into denom (zeroed)
__global__ void k_node_self(const float* as_, const float* ad_, const float* lattr,
                            const float* kc, int kcoff, const unsigned* amax_enc,
                            float* amaxf, float* aself, float* denom, int n, int heads) {
  int idx = blockIdx.x * blockDim.x + threadIdx.x;
  if (idx >= n * heads) return;
  int node = idx / heads, hd = idx % heads;
  float als = lrelu(as_[idx] + ad_[idx] + lattr[node] * kc[kcoff + hd]);
  float am = fmaxf(dec_f(amax_enc[idx]), als);  // dec(0)=NaN -> fmaxf picks als
  amaxf[idx] = am;
  aself[idx] = als;
  denom[idx] = expf(als - am);  // runs before edge atomicAdds (stream order)
}

__global__ void k_edge_exp(const int* __restrict__ dst, const float* __restrict__ amaxf,
                           float* alpha_ex, float* denom, int E, int heads) {
  int idx = blockIdx.x * blockDim.x + threadIdx.x;
  if (idx >= E * heads) return;
  int e = idx / heads, hd = idx % heads;
  float ex = expf(alpha_ex[idx] - amaxf[dst[e] * heads + hd]);
  alpha_ex[idx] = ex;
  atomicAdd(&denom[dst[e] * heads + hd], ex);
}

// one wave per edge, 4 channels/lane (128 features, 2 heads)
__global__ void k_edge_aggr128(const int* __restrict__ src, const int* __restrict__ dst,
                               const float* __restrict__ ex, const float* __restrict__ denom,
                               const float* __restrict__ h, float* out, int E) {
  int t = blockIdx.x * blockDim.x + threadIdx.x;
  int e = t >> 5, lane = t & 31;
  if (e >= E) return;
  int s = src[e], d = dst[e];
  int c0 = lane * 4, hd = c0 >> 6;
  float coef = ex[e * 2 + hd] / (denom[d * 2 + hd] + 1e-16f);
  const float4 hv = *reinterpret_cast<const float4*>(h + (size_t)s * F1 + c0);
  float* o = out + (size_t)d * F1 + c0;
  atomicAdd(o + 0, hv.x * coef);
  atomicAdd(o + 1, hv.y * coef);
  atomicAdd(o + 2, hv.z * coef);
  atomicAdd(o + 3, hv.w * coef);
}

// one wave per edge, 1 channel/lane (32 features, 1 head)
__global__ void k_edge_aggr32(const int* __restrict__ src, const int* __restrict__ dst,
                              const float* __restrict__ ex, const float* __restrict__ denom,
                              const float* __restrict__ h, float* out, int E) {
  int t = blockIdx.x * blockDim.x + threadIdx.x;
  int e = t >> 5, lane = t & 31;
  if (e >= E) return;
  int s = src[e], d = dst[e];
  float coef = ex[e] / (denom[d] + 1e-16f);
  atomicAdd(out + (size_t)d * F2 + lane, h[(size_t)s * F2 + lane] * coef);
}

// layer-1 epilogue: add self-loop term + bias, apply ELU (in place on aggregated buf)
__global__ void k_final1(const float* __restrict__ h1, const float* aself,
                         const float* amaxf, const float* denom,
                         const float* bias, float* inout, int n) {
  int idx = blockIdx.x * blockDim.x + threadIdx.x;
  if (idx >= n * F1) return;
  int node = idx >> 7, c = idx & 127, hd = c >> 6;
  int a = node * H1 + hd;
  float coef = expf(aself[a] - amaxf[a]) / (denom[a] + 1e-16f);
  float v = inout[idx] + h1[idx] * coef + bias[c];
  inout[idx] = v > 0.f ? v : expm1f(v);
}

// layer-2 epilogue: add self-loop term + bias (no activation)
__global__ void k_final2(const float* __restrict__ h2, const float* aself,
                         const float* amaxf, const float* denom,
                         const float* bias, float* out, int n) {
  int idx = blockIdx.x * blockDim.x + threadIdx.x;
  if (idx >= n * F2) return;
  int node = idx >> 5, c = idx & 31;
  float coef = expf(aself[node] - amaxf[node]) / (denom[node] + 1e-16f);
  out[idx] += h2[idx] * coef + bias[c];
}

// ---------------- launch ----------------
extern "C" void kernel_launch(void* const* d_in, const int* in_sizes, int n_in,
                              void* d_out, int out_size, void* d_ws, size_t ws_size,
                              hipStream_t stream) {
  const float* x   = (const float*)d_in[0];
  const int*   ei  = (const int*)d_in[1];
  const int*   src = ei;
  const int*   dst = ei + NE;
  const float* ew  = (const float*)d_in[2];
  const float* W1  = (const float*)d_in[3];
  const float* as1 = (const float*)d_in[4];
  const float* ad1 = (const float*)d_in[5];
  const float* We1 = (const float*)d_in[6];
  const float* ae1 = (const float*)d_in[7];
  const float* b1  = (const float*)d_in[8];
  const float* W2  = (const float*)d_in[9];
  const float* as2 = (const float*)d_in[10];
  const float* ad2 = (const float*)d_in[11];
  const float* We2 = (const float*)d_in[12];
  const float* ae2 = (const float*)d_in[13];
  const float* b2  = (const float*)d_in[14];
  float* out = (float*)d_out;

  // workspace layout (bytes)
  char* ws = (char*)d_ws;
  float*    h1      = (float*)(ws + 0);              // N*128  (layer2: h2 = first N*32)
  float*    hmid    = (float*)(ws + 25600000);       // N*128  (agg1 -> elu -> layer2 input)
  float*    exbuf   = (float*)(ws + 51200000);       // E*2
  float*    asrc    = (float*)(ws + 57600000);       // N*2
  float*    adst    = (float*)(ws + 58000000);       // N*2
  float*    aself   = (float*)(ws + 58400000);       // N*2
  unsigned* amaxenc = (unsigned*)(ws + 58800000);    // N*2
  float*    amaxf   = (float*)(ws + 59200000);       // N*2
  float*    denom   = (float*)(ws + 59600000);       // N*2
  float*    deg     = (float*)(ws + 60000000);       // N
  float*    lattr   = (float*)(ws + 60200000);       // N
  float*    kc      = (float*)(ws + 60400000);       // 3 consts
  float*    h2      = h1;                            // overlay (h1 dead by then)

  auto cdiv = [](int a, int b) { return (a + b - 1) / b; };

  // -------- shared preprocessing --------
  hipMemsetAsync(deg, 0, NN * 4, stream);
  hipMemsetAsync(lattr, 0, NN * 4, stream);
  k_deg_loop<<<cdiv(NE, 256), 256, 0, stream>>>(src, dst, ew, deg, lattr, NE);
  k_loop_attr<<<cdiv(NN, 256), 256, 0, stream>>>(lattr, deg, NN);
  k_consts<<<1, 1, 0, stream>>>(We1, ae1, We2, ae2, kc);

  // -------- layer 1 --------
  k_gemm_bf16<INF1, F1><<<cdiv(NN, 32), 256, 0, stream>>>(x, W1, h1, NN);   // 8 waves: 128 cols
  k_node_attn<<<cdiv(NN * H1, 256), 256, 0, stream>>>(h1, as1, ad1, asrc, adst, NN, H1, C1);
  hipMemsetAsync(amaxenc, 0, NN * H1 * 4, stream);
  hipMemsetAsync(denom, 0, NN * H1 * 4, stream);
  k_edge_alpha_max<<<cdiv(NE * H1, 256), 256, 0, stream>>>(src, dst, ew, asrc, adst, kc, 0,
                                                           exbuf, amaxenc, NE, H1);
  k_node_self<<<cdiv(NN * H1, 256), 256, 0, stream>>>(asrc, adst, lattr, kc, 0, amaxenc,
                                                      amaxf, aself, denom, NN, H1);
  k_edge_exp<<<cdiv(NE * H1, 256), 256, 0, stream>>>(dst, amaxf, exbuf, denom, NE, H1);
  hipMemsetAsync(hmid, 0, (size_t)NN * F1 * 4, stream);
  k_edge_aggr128<<<cdiv(NE * 32, 256), 256, 0, stream>>>(src, dst, exbuf, denom, h1, hmid, NE);
  k_final1<<<cdiv(NN * F1, 256), 256, 0, stream>>>(h1, aself, amaxf, denom, b1, hmid, NN);

  // -------- layer 2 --------
  k_gemm_bf16<F1, F2><<<cdiv(NN, 32), 64, 0, stream>>>(hmid, W2, h2, NN);   // 2 waves: 32 cols
  k_node_attn<<<cdiv(NN, 256), 256, 0, stream>>>(h2, as2, ad2, asrc, adst, NN, 1, F2);
  hipMemsetAsync(amaxenc, 0, NN * 4, stream);
  hipMemsetAsync(denom, 0, NN * 4, stream);
  k_edge_alpha_max<<<cdiv(NE, 256), 256, 0, stream>>>(src, dst, ew, asrc, adst, kc, 2,
                                                      exbuf, amaxenc, NE, 1);
  k_node_self<<<cdiv(NN, 256), 256, 0, stream>>>(asrc, adst, lattr, kc, 2, amaxenc,
                                                 amaxf, aself, denom, NN, 1);
  k_edge_exp<<<cdiv(NE, 256), 256, 0, stream>>>(dst, amaxf, exbuf, denom, NE, 1);
  hipMemsetAsync(out, 0, (size_t)NN * F2 * 4, stream);
  k_edge_aggr32<<<cdiv(NE * 32, 256), 256, 0, stream>>>(src, dst, exbuf, denom, h2, out, NE);
  k_final2<<<cdiv(NN * F2, 256), 256, 0, stream>>>(h2, aself, amaxf, denom, b2, out, NN);
}
